// APGLinear_83468394430799
// MI455X (gfx1250) — compile-verified, matrix operand order checked by reading, not run on hardware
//
#include <hip/hip_runtime.h>
#include <hip/hip_bf16.h>

// ---------------- problem constants ----------------
#define BATCH     4096
#define IN_DIM    256
#define OUT_DIM   256
#define COND      128
#define HID       512
#define HYP_COLS  513          // HID + constant-1 column (folds b2 into the GEMM)
#define HYP_STRIDE 520         // padded row stride (ushorts), 16B aligned
#define K_TOTAL   (HYP_COLS * IN_DIM)   // 131328
#define K_STEP    64
#define NSTEPS    (K_TOTAL / K_STEP)    // 2052
#define KSPLIT    2
#define NSTEPS_H  (NSTEPS / KSPLIT)     // 1026 steps per K-half
#define BLK_M     64
#define BLK_N     128
#define PADF      260          // inLf row stride in floats (256 + 4 pad)
#define PF_DIST   4            // prefetch distance in K steps

typedef __attribute__((ext_vector_type(16))) __bf16 v16bf;
typedef __attribute__((ext_vector_type(8)))  float  v8f;

__device__ __forceinline__ unsigned short f2bf(float f) {
    union { float f; unsigned u; } x; x.f = f;
    unsigned u = x.u;
    u += 0x7fffu + ((u >> 16) & 1u);   // round-to-nearest-even
    return (unsigned short)(u >> 16);
}
// RNE pack: two floats -> dword of two bf16 (lo in [15:0], hi in [31:16])
__device__ __forceinline__ unsigned pack2bf(float lo, float hi) {
    union { float f; unsigned u; } a, b; a.f = lo; b.f = hi;
    unsigned ua = a.u + (0x7fffu + ((a.u >> 16) & 1u));
    unsigned ub = b.u + (0x7fffu + ((b.u >> 16) & 1u));
    return (ua >> 16) | (ub & 0xffff0000u);
}
// fast round-half-up pack: 2x v_add + 1x v_perm (differs from RNE only on exact ties)
__device__ __forceinline__ unsigned packrhu(float lo, float hi) {
    union { float f; unsigned u; } a, b; a.f = lo; b.f = hi;
    return __builtin_amdgcn_perm(b.u + 0x8000u, a.u + 0x8000u, 0x07060302u);
}
__device__ __forceinline__ float bits2f(unsigned u) {
    union { unsigned u; float f; } x; x.u = u; return x.f;
}
__device__ __forceinline__ v16bf mkfrag(uint4 lo, uint4 hi) {
    union { uint4 q[2]; v16bf v; } c; c.q[0] = lo; c.q[1] = hi; return c.v;
}

// ---------------- kernel 1: hyp = relu(z @ W1 + b1), bf16, +constant-1 col ----------------
__global__ void hyp_kernel(const float* __restrict__ z, const float* __restrict__ W1,
                           const float* __restrict__ b1, unsigned short* __restrict__ hyp) {
    __shared__ float zl[COND];
    const int b = blockIdx.x;
    const int t = threadIdx.x;
    if (t < COND) zl[t] = z[b * COND + t];
    __syncthreads();
    for (int h = t; h < HID; h += 256) {
        float acc = b1[h];
#pragma unroll 8
        for (int c = 0; c < COND; ++c) acc += zl[c] * W1[c * HID + h];
        acc = acc > 0.f ? acc : 0.f;
        hyp[(long)b * HYP_STRIDE + h] = f2bf(acc);
    }
    if (t == 0) hyp[(long)b * HYP_STRIDE + HID] = 0x3F80;  // 1.0 in bf16
}

// ---------------- kernel 2: Bmat[k,o] = bf16( W2 ‖ b2 )  (identity linear reindex, RNE) ----------------
__global__ void convert_kernel(const float* __restrict__ W2, const float* __restrict__ b2,
                               unsigned short* __restrict__ Bmat) {
    const long i4   = (long)blockIdx.x * blockDim.x + threadIdx.x;
    const long flat = i4 * 4;
    const long W2N  = (long)HID * IN_DIM * OUT_DIM;  // 33,554,432
    float4 v;
    if (flat < W2N) v = *(const float4*)(W2 + flat);
    else            v = *(const float4*)(b2 + (flat - W2N));
    uint2 o;
    o.x = pack2bf(v.x, v.y);
    o.y = pack2bf(v.z, v.w);
    *(uint2*)(Bmat + flat) = o;
}

// ---------------- kernel 3: fused outer-product GEMM with bf16 WMMA (split-K) ----------------
// part[ks][4096,256] = A[4096, Kslice] * Bmat[Kslice, 256],
// A[b,k] = hyp[b,k>>8] * input[b,k&255]
__global__ __launch_bounds__(256) void gemm_kernel(
    const float* __restrict__ input_h,
    const unsigned short* __restrict__ hyp,    // [4096][HYP_STRIDE] bf16
    const unsigned short* __restrict__ Bmat,   // [K_TOTAL][256] bf16
    float* __restrict__ part)                  // [KSPLIT][4096][256]
{
    // LDS fragment images are stored lane-linear: [lane][16 ushorts] == exactly the
    // per-lane v16bf WMMA operand (A 16x32 layout: lane = m + 16*((k>>3)&1),
    // idx = (k&7) + 8*(k>>4); B 32x16 mirrored with n in place of m).
    __shared__ alignas(16) float          inLf[BLK_M][PADF];           // input tile, f32
    __shared__ alignas(32) unsigned short Aimg[2][2][4][32][16];       // [buf][kb2][mt][lane][idx]
    __shared__ alignas(32) unsigned short Bimg[2][2][8][32][16];       // [buf][kb2][nt][lane][idx]

    const int t    = threadIdx.x;
    const int bm   = blockIdx.x & 63;
    const int bn   = (blockIdx.x >> 6) & 1;
    const int ks   = blockIdx.x >> 7;          // K-split index 0..1
    const int b0   = bm * BLK_M;
    const int n0   = bn * BLK_N;
    const int s_lo = ks * NSTEPS_H;
    const int s_hi = s_lo + NSTEPS_H;
    const int lane = t & 31;
    const int wave = t >> 5;
    const int mt   = wave & 3;   // wave's M tile (16 rows)
    const int ng   = wave >> 2;  // wave's N group (4 tiles = 64 cols)

    // ---- prologue: async DMA the input tile (f32) global -> LDS (ASYNCcnt path) ----
#pragma unroll
    for (int r = 0; r < 16; ++r) {
        int e   = t + 256 * r;        // float4 chunk index, 0..4095
        int row = e >> 6;
        int c4  = e & 63;
        const void* gsrc = (const void*)(input_h + (long)(b0 + row) * IN_DIM + c4 * 4);
        unsigned    ldst = (unsigned)(unsigned long long)&inLf[row][c4 * 4];
        asm volatile("global_load_async_to_lds_b128 %0, %1, off"
                     :: "v"(ldst), "v"(gsrc) : "memory");
    }
    asm volatile("s_wait_asynccnt 0x0" ::: "memory");

    // per-thread staging registers (pipelined across iterations)
    unsigned int   br[8][4];       // 8 B-rows x 4 dwords (threads 0..127)
    unsigned short hypreg[4];      // prefetched hyp scalars (threads 128..255)

    auto issueB = [&](int s) {     // coalesced b128 row loads for next step
        int k0 = s * K_STEP;
        int kg = t >> 4, ngrp = t & 15;
        const unsigned short* src = Bmat + (long)(k0 + kg * 8) * OUT_DIM + n0 + ngrp * 8;
#pragma unroll
        for (int rr = 0; rr < 8; ++rr) {
            uint4 v = *(const uint4*)(src + rr * OUT_DIM);
            br[rr][0] = v.x; br[rr][1] = v.y; br[rr][2] = v.z; br[rr][3] = v.w;
        }
        // pull L2-resident W2 lines toward the WGP ahead of time (global_prefetch_b8)
        if (s + PF_DIST < s_hi)
            __builtin_prefetch(src + (long)PF_DIST * K_STEP * OUT_DIM, 0, 3);
    };
    auto issueA = [&](int s) {
        int h  = (s * K_STEP) >> 8;
        int tt = t - 128;
#pragma unroll
        for (int rep = 0; rep < 4; ++rep) {
            int bb = (tt >> 3) + 16 * rep;
            hypreg[rep] = hyp[(long)(b0 + bb) * HYP_STRIDE + h];
        }
    };

    auto storeB = [&](int buf) {   // 8x8 register transpose -> fragment-layout LDS
        int kg = t >> 4, ngrp = t & 15;
        int kb2 = kg >> 2, half = kg & 1, idxb = (kg >> 1) & 1;
        int nt = ngrp >> 1, nlbase = (ngrp & 1) * 8;
#pragma unroll
        for (int j = 0; j < 4; ++j) {
#pragma unroll
            for (int sub = 0; sub < 2; ++sub) {
                int nl = j * 2 + sub;
                unsigned sel = sub ? 0x07060302u : 0x05040100u;
                uint4 ov;
                ov.x = __builtin_amdgcn_perm(br[1][j], br[0][j], sel);
                ov.y = __builtin_amdgcn_perm(br[3][j], br[2][j], sel);
                ov.z = __builtin_amdgcn_perm(br[5][j], br[4][j], sel);
                ov.w = __builtin_amdgcn_perm(br[7][j], br[6][j], sel);
                *(uint4*)&Bimg[buf][kb2][nt][nlbase + nl + 16 * half][idxb * 8] = ov;
            }
        }
    };
    auto storeA = [&](int buf, int s) {  // generate A = hyp*input, fast bf16 pack
        int i0 = (s * K_STEP) & 255;
        int tt = t - 128;
        int kg = tt & 7;
        int lm = tt >> 3;                              // 0..15; bb = lm + 16*rep => mtt == rep
        int kb2 = kg >> 2, half = kg & 1, idxb = (kg >> 1) & 1;
        const float* ipb = &inLf[lm][i0 + kg * 8];
        float4 pa[4], pb[4];
#pragma unroll
        for (int rep = 0; rep < 4; ++rep) {            // batch all LDS reads first
            pa[rep] = *(const float4*)(ipb + rep * 16 * PADF);
            pb[rep] = *(const float4*)(ipb + rep * 16 * PADF + 4);
        }
        unsigned short* dst = &Aimg[buf][kb2][0][lm + 16 * half][idxb * 8];
#pragma unroll
        for (int rep = 0; rep < 4; ++rep) {
            float hv = bits2f(((unsigned)hypreg[rep]) << 16);
            uint4 ov;
            ov.x = packrhu(hv * pa[rep].x, hv * pa[rep].y);
            ov.y = packrhu(hv * pa[rep].z, hv * pa[rep].w);
            ov.z = packrhu(hv * pb[rep].x, hv * pb[rep].y);
            ov.w = packrhu(hv * pb[rep].z, hv * pb[rep].w);
            *(uint4*)(dst + rep * 512) = ov;           // mt stride = 32*16 ushorts
        }
    };

    v8f acc[4];
#pragma unroll
    for (int j = 0; j < 4; ++j) acc[j] = (v8f){0.f, 0.f, 0.f, 0.f, 0.f, 0.f, 0.f, 0.f};

    if (t < 128) issueB(s_lo); else issueA(s_lo);
    __syncthreads();   // inLf (async) visible before first storeA

#pragma unroll 1
    for (int s = s_lo; s < s_hi; ++s) {
        const int buf = s & 1;
        if (t < 128) storeB(buf); else storeA(buf, s);
        __syncthreads();
        if (s + 1 < s_hi) { if (t < 128) issueB(s + 1); else issueA(s + 1); }

        // ---- compute: ONE asm block = 20 batched ds_load_b128 + ONE s_wait_dscnt.
        // WMMAs data-depend on the block's outputs, so they cannot start before the
        // wait; the waitcnt pass cannot see through inline asm, so the wait lives
        // inside the block to stay correct on hardware.
        unsigned addrA = (unsigned)(unsigned long long)&Aimg[buf][0][mt][lane][0];
        unsigned addrB = (unsigned)(unsigned long long)&Bimg[buf][0][ng * 4][lane][0];
        uint4 a0l, a0h, a1l, a1h;
        uint4 b0l, b0h, b1l, b1h, b2l, b2h, b3l, b3h;
        uint4 b4l, b4h, b5l, b5h, b6l, b6h, b7l, b7h;
        asm volatile(
            "ds_load_b128 %0, %20\n\t"
            "ds_load_b128 %1, %20 offset:16\n\t"
            "ds_load_b128 %2, %20 offset:4096\n\t"
            "ds_load_b128 %3, %20 offset:4112\n\t"
            "ds_load_b128 %4, %21\n\t"
            "ds_load_b128 %5, %21 offset:16\n\t"
            "ds_load_b128 %6, %21 offset:1024\n\t"
            "ds_load_b128 %7, %21 offset:1040\n\t"
            "ds_load_b128 %8, %21 offset:2048\n\t"
            "ds_load_b128 %9, %21 offset:2064\n\t"
            "ds_load_b128 %10, %21 offset:3072\n\t"
            "ds_load_b128 %11, %21 offset:3088\n\t"
            "ds_load_b128 %12, %21 offset:8192\n\t"
            "ds_load_b128 %13, %21 offset:8208\n\t"
            "ds_load_b128 %14, %21 offset:9216\n\t"
            "ds_load_b128 %15, %21 offset:9232\n\t"
            "ds_load_b128 %16, %21 offset:10240\n\t"
            "ds_load_b128 %17, %21 offset:10256\n\t"
            "ds_load_b128 %18, %21 offset:11264\n\t"
            "ds_load_b128 %19, %21 offset:11280\n\t"
            "s_wait_dscnt 0x0"
            : "=v"(a0l), "=v"(a0h), "=v"(a1l), "=v"(a1h),
              "=v"(b0l), "=v"(b0h), "=v"(b1l), "=v"(b1h),
              "=v"(b2l), "=v"(b2h), "=v"(b3l), "=v"(b3h),
              "=v"(b4l), "=v"(b4h), "=v"(b5l), "=v"(b5h),
              "=v"(b6l), "=v"(b6h), "=v"(b7l), "=v"(b7h)
            : "v"(addrA), "v"(addrB)
            : "memory");
        v16bf av0 = mkfrag(a0l, a0h);
        v16bf av1 = mkfrag(a1l, a1h);
        v16bf bv0 = mkfrag(b0l, b0h);
        v16bf bv1 = mkfrag(b1l, b1h);
        v16bf bv2 = mkfrag(b2l, b2h);
        v16bf bv3 = mkfrag(b3l, b3h);
        v16bf bv4 = mkfrag(b4l, b4h);
        v16bf bv5 = mkfrag(b5l, b5h);
        v16bf bv6 = mkfrag(b6l, b6h);
        v16bf bv7 = mkfrag(b7l, b7h);
        acc[0] = __builtin_amdgcn_wmma_f32_16x16x32_bf16(false, av0, false, bv0, (short)0, acc[0], false, false);
        acc[1] = __builtin_amdgcn_wmma_f32_16x16x32_bf16(false, av0, false, bv1, (short)0, acc[1], false, false);
        acc[2] = __builtin_amdgcn_wmma_f32_16x16x32_bf16(false, av0, false, bv2, (short)0, acc[2], false, false);
        acc[3] = __builtin_amdgcn_wmma_f32_16x16x32_bf16(false, av0, false, bv3, (short)0, acc[3], false, false);
        acc[0] = __builtin_amdgcn_wmma_f32_16x16x32_bf16(false, av1, false, bv4, (short)0, acc[0], false, false);
        acc[1] = __builtin_amdgcn_wmma_f32_16x16x32_bf16(false, av1, false, bv5, (short)0, acc[1], false, false);
        acc[2] = __builtin_amdgcn_wmma_f32_16x16x32_bf16(false, av1, false, bv6, (short)0, acc[2], false, false);
        acc[3] = __builtin_amdgcn_wmma_f32_16x16x32_bf16(false, av1, false, bv7, (short)0, acc[3], false, false);
    }

    // ---- epilogue: C layout lane = n + 16*(m>>3), vgpr v = m&7; plain stores (deterministic) ----
    const int rowbase = b0 + mt * 16 + (lane >> 4) * 8;
    const int colbase = n0 + ng * 64 + (lane & 15);
    float* pout = part + ((long)ks << 20);   // 4096*256 = 1<<20 per K-half
#pragma unroll
    for (int j = 0; j < 4; ++j) {
        int col = colbase + j * 16;
#pragma unroll
        for (int v = 0; v < 8; ++v)
            pout[(long)(rowbase + v) * OUT_DIM + col] = acc[j][v];
    }
}

// ---------------- kernel 4: out = part0 + part1 + bias ----------------
__global__ void reduce_kernel(const float* __restrict__ part, const float* __restrict__ bias,
                              float* __restrict__ out) {
    const int i = blockIdx.x * 256 + threadIdx.x;   // 0 .. 2^20-1
    out[i] = part[i] + part[i + (1 << 20)] + bias[i & (OUT_DIM - 1)];
}

// ---------------- host-side launch ----------------
extern "C" void kernel_launch(void* const* d_in, const int* in_sizes, int n_in,
                              void* d_out, int out_size, void* d_ws, size_t ws_size,
                              hipStream_t stream) {
    (void)in_sizes; (void)n_in; (void)out_size; (void)ws_size;
    const float* input_h = (const float*)d_in[0];
    const float* z       = (const float*)d_in[1];
    const float* W1      = (const float*)d_in[2];
    const float* b1      = (const float*)d_in[3];
    const float* W2      = (const float*)d_in[4];
    const float* b2      = (const float*)d_in[5];
    const float* bias    = (const float*)d_in[6];
    float* out           = (float*)d_out;

    unsigned short* hyp  = (unsigned short*)d_ws;                 // 4096 x 520 bf16
    unsigned short* Bmat = hyp + (long)BATCH * HYP_STRIDE;        // 131328 x 256 bf16
    float*          part = (float*)(Bmat + (long)K_TOTAL * OUT_DIM);  // 2 x 4096 x 256 f32

    hyp_kernel<<<BATCH, 256, 0, stream>>>(z, W1, b1, hyp);

    const long totalB = (long)K_TOTAL * OUT_DIM;                  // 33,619,968
    convert_kernel<<<(unsigned)(totalB / 4 / 256), 256, 0, stream>>>(W2, b2, Bmat);

    gemm_kernel<<<(BATCH / BLK_M) * (OUT_DIM / BLK_N) * KSPLIT, 256, 0, stream>>>(
        input_h, hyp, Bmat, part);

    reduce_kernel<<<(BATCH * OUT_DIM) / 256, 256, 0, stream>>>(part, bias, out);
}
